// LSTM_33406255628730
// MI455X (gfx1250) — compile-verified, matrix operand order checked by reading, not run on hardware
//
#include <hip/hip_runtime.h>

// ---------------------------------------------------------------------------
// LSTM on MI455X (gfx1250): bf16 WMMA (16x16x32) with f32 accumulation.
// v3: software-pipelined fragment streams (register double-buffering) so
// WMMAs overlap the L2 loads; Wh staged via double-buffered
// global_load_async_to_lds_b128 (ASYNCcnt-gated), read back with ds_load_b128.
// ---------------------------------------------------------------------------

constexpr int T  = 256;
constexpr int D  = 512;
constexpr int H  = 1024;
constexpr int NC = 128;
constexpr int B  = 512;
constexpr int NG = 4 * H;          // 4096 fused gate columns (i,f,g,o)
constexpr int BT = B * T;          // 131072 rows for input projection

// LDS staging geometry for the recurrent step kernel.
constexpr int KC    = 64;          // k-chunk depth staged per buffer
constexpr int RSB   = KC + 8;      // padded row stride (elements) = 144 B
constexpr int BROWS = 256;         // 4 gates * 64 hidden cols per block
constexpr size_t STEP_LDS = 2ull * BROWS * RSB * sizeof(unsigned short); // 73728 B

typedef __attribute__((ext_vector_type(16))) __bf16        v16bf;
typedef __attribute__((ext_vector_type(8)))  float         v8f;
typedef __attribute__((ext_vector_type(4)))  unsigned int  u32x4;
typedef __attribute__((ext_vector_type(8)))  unsigned int  u32x8;
typedef __attribute__((ext_vector_type(4)))  float         f32x4;
typedef __attribute__((ext_vector_type(4)))  unsigned short u16x4;

// ---- bf16 helpers (storage type: unsigned short) ---------------------------
__device__ __forceinline__ unsigned short f2bf(float f) {
  unsigned int u = __builtin_bit_cast(unsigned int, f);
  u += 0x7FFFu + ((u >> 16) & 1u);            // round to nearest even
  return (unsigned short)(u >> 16);
}
__device__ __forceinline__ float bf2f(unsigned short h) {
  unsigned int u = ((unsigned int)h) << 16;
  return __builtin_bit_cast(float, u);
}
__device__ __forceinline__ float sigmoid_f(float x) {
  return 1.0f / (1.0f + __expf(-x));
}
__device__ __forceinline__ float tanh_f(float x) {
  float e = __expf(2.0f * x);                 // inf-safe: -> 1.0
  return 1.0f - 2.0f / (1.0f + e);
}

__device__ __forceinline__ v16bf pack_frag(u32x4 lo, u32x4 hi) {
  u32x8 r;
  r[0] = lo[0]; r[1] = lo[1]; r[2] = lo[2]; r[3] = lo[3];
  r[4] = hi[0]; r[5] = hi[1]; r[6] = hi[2]; r[7] = hi[3];
  return __builtin_bit_cast(v16bf, r);
}

// ---- WMMA fragment load from global (row-major bf16 [rows, ld]) ------------
// ISA 16-bit A/B layout: lanes 0-15: row0+lane, K k0+0..7 & k0+16..23;
//                        lanes16-31: row0+lane-16, K k0+8..15 & k0+24..31.
__device__ __forceinline__ v16bf load_tile_bf16(const unsigned short* __restrict__ base,
                                                int row0, int ld, int k0) {
  const int lane = threadIdx.x & 31;
  const unsigned short* p =
      base + (size_t)(row0 + (lane & 15)) * (size_t)ld + (size_t)(k0 + ((lane >> 4) << 3));
  u32x4 lo = *(const u32x4*)p;
  u32x4 hi = *(const u32x4*)(p + 16);
  return pack_frag(lo, hi);
}

__device__ __forceinline__ v8f wmma_bf16(v16bf a, v16bf b, v8f c) {
  // (neg_a, A, neg_b, B, c_mod, C, reuse_a, reuse_b)
  return __builtin_amdgcn_wmma_f32_16x16x32_bf16(false, a, false, b, (short)0, c,
                                                 false, false);
}

// ---------------------------------------------------------------------------
// Kernel 1: pack weights to bf16, transposed to [N, K] row-major.
// ---------------------------------------------------------------------------
__global__ void pack_weights_kernel(
    const float* __restrict__ Wix, const float* __restrict__ Wfx,
    const float* __restrict__ Wgx, const float* __restrict__ Wox,
    const float* __restrict__ Wih, const float* __restrict__ Wfh,
    const float* __restrict__ Wgh, const float* __restrict__ Woh,
    const float* __restrict__ bi,  const float* __restrict__ bfv,
    const float* __restrict__ bg,  const float* __restrict__ bo,
    const float* __restrict__ Wph,
    unsigned short* __restrict__ WxT, unsigned short* __restrict__ WhT,
    unsigned short* __restrict__ WpT, float* __restrict__ bias) {
  const long NWX = (long)NG * D;
  const long NWH = (long)NG * H;
  const long NWP = (long)NC * H;
  const long total = NWX + NWH + NWP + NG;
  long i = (long)blockIdx.x * blockDim.x + threadIdx.x;
  const long stride = (long)gridDim.x * blockDim.x;
  for (; i < total; i += stride) {
    if (i < NWX) {                                   // WxT[n,k], k<D
      long n = i / D, k = i % D;
      int g = (int)(n >> 10), nh = (int)(n & (H - 1));
      const float* W = (g == 0) ? Wix : (g == 1) ? Wfx : (g == 2) ? Wgx : Wox;
      WxT[i] = f2bf(W[k * H + nh]);
    } else if (i < NWX + NWH) {                      // WhT[n,k], k<H
      long j = i - NWX;
      long n = j / H, k = j % H;
      int g = (int)(n >> 10), nh = (int)(n & (H - 1));
      const float* W = (g == 0) ? Wih : (g == 1) ? Wfh : (g == 2) ? Wgh : Woh;
      WhT[j] = f2bf(W[k * H + nh]);
    } else if (i < NWX + NWH + NWP) {                // WpT[n,k]: Wph transposed
      long j = i - NWX - NWH;
      long n = j / H, k = j % H;
      WpT[j] = f2bf(Wph[k * NC + n]);
    } else {                                         // fused bias
      long n = i - NWX - NWH - NWP;
      int g = (int)(n >> 10), nh = (int)(n & (H - 1));
      const float* bv = (g == 0) ? bi : (g == 1) ? bfv : (g == 2) ? bg : bo;
      bias[n] = bv[nh];
    }
  }
}

// ---------------------------------------------------------------------------
// Kernel 2: x f32 -> bf16 (layout [B*T, D] == flat x layout).
// ---------------------------------------------------------------------------
__global__ void pack_x_kernel(const float* __restrict__ x,
                              unsigned short* __restrict__ xb, long n4) {
  long i = (long)blockIdx.x * blockDim.x + threadIdx.x;
  const long stride = (long)gridDim.x * blockDim.x;
  for (; i < n4; i += stride) {
    f32x4 v = ((const f32x4*)x)[i];
    u16x4 o;
    o[0] = f2bf(v[0]); o[1] = f2bf(v[1]); o[2] = f2bf(v[2]); o[3] = f2bf(v[3]);
    ((u16x4*)xb)[i] = o;
  }
}

// ---------------------------------------------------------------------------
// Kernel 3: zero c and h ping buffer (workspace is poisoned by harness).
// ---------------------------------------------------------------------------
__global__ void init_state_kernel(float* __restrict__ c,
                                  unsigned short* __restrict__ hb0, int n) {
  int i = blockIdx.x * blockDim.x + threadIdx.x;
  const int stride = gridDim.x * blockDim.x;
  for (; i < n; i += stride) { c[i] = 0.0f; hb0[i] = 0; }
}

// ---------------------------------------------------------------------------
// Kernel 4: input projection GEMM, software-pipelined: loads for k-step k+32
// are issued before the WMMAs of k-step k (peeled final step).
// ---------------------------------------------------------------------------
__global__ void __launch_bounds__(128)
input_gemm_kernel(const unsigned short* __restrict__ xb,
                  const unsigned short* __restrict__ WxT,
                  const float* __restrict__ bias,
                  unsigned short* __restrict__ xprojb) {
  const int wave = threadIdx.x >> 5;
  const int r0   = blockIdx.x * 16;
  const int nh0  = (blockIdx.y * 4 + wave) * 16;
  v8f a0 = {0.f,0.f,0.f,0.f,0.f,0.f,0.f,0.f};
  v8f a1 = a0, a2 = a0, a3 = a0;

  v16bf a  = load_tile_bf16(xb,  r0,           D, 0);
  v16bf b0 = load_tile_bf16(WxT, 0 * H + nh0,  D, 0);
  v16bf b1 = load_tile_bf16(WxT, 1 * H + nh0,  D, 0);
  v16bf b2 = load_tile_bf16(WxT, 2 * H + nh0,  D, 0);
  v16bf b3 = load_tile_bf16(WxT, 3 * H + nh0,  D, 0);
  for (int k0 = 32; k0 < D; k0 += 32) {
    v16bf an  = load_tile_bf16(xb,  r0,          D, k0);
    v16bf b0n = load_tile_bf16(WxT, 0 * H + nh0, D, k0);
    v16bf b1n = load_tile_bf16(WxT, 1 * H + nh0, D, k0);
    v16bf b2n = load_tile_bf16(WxT, 2 * H + nh0, D, k0);
    v16bf b3n = load_tile_bf16(WxT, 3 * H + nh0, D, k0);
    a0 = wmma_bf16(a, b0, a0);
    a1 = wmma_bf16(a, b1, a1);
    a2 = wmma_bf16(a, b2, a2);
    a3 = wmma_bf16(a, b3, a3);
    a = an; b0 = b0n; b1 = b1n; b2 = b2n; b3 = b3n;
  }
  a0 = wmma_bf16(a, b0, a0);
  a1 = wmma_bf16(a, b1, a1);
  a2 = wmma_bf16(a, b2, a2);
  a3 = wmma_bf16(a, b3, a3);

  const int lane = threadIdx.x & 31;
  const int ncol = nh0 + (lane & 15);
  const int mb   = (lane >> 4) * 8;
#pragma unroll
  for (int v = 0; v < 8; ++v) {
    const int r = r0 + mb + v;          // row in [0, B*T)
    const int b_ = r / T, t = r % T;    // x is [B, T, D]
    const size_t o = ((size_t)t * B + b_) * NG;
    xprojb[o + 0 * H + ncol] = f2bf(a0[v] + bias[0 * H + ncol]);
    xprojb[o + 1 * H + ncol] = f2bf(a1[v] + bias[1 * H + ncol]);
    xprojb[o + 2 * H + ncol] = f2bf(a2[v] + bias[2 * H + ncol]);
    xprojb[o + 3 * H + ncol] = f2bf(a3[v] + bias[3 * H + ncol]);
  }
}

// ---------------------------------------------------------------------------
// Kernel 5: one recurrent step.
// Block = 4 waves. Block output: 32 batch rows x 64 hidden cols. Wave owns one
// 16-col tile for all 4 gates and both M-tiles (8 f32 accumulators). Wh slabs
// are staged wave-privately into a double-buffered LDS region with
// global_load_async_to_lds_b128; A fragments are register-pipelined from L2.
// ---------------------------------------------------------------------------
__global__ void __launch_bounds__(128)
lstm_step_kernel(const unsigned short* __restrict__ WhT,
                 const unsigned short* __restrict__ xprojb,
                 const unsigned short* __restrict__ hb_in,
                 unsigned short* __restrict__ hb_out,
                 float* __restrict__ c, int t) {
  extern __shared__ unsigned short smem[];          // 2 buffers * BROWS * RSB
  const int wave = threadIdx.x >> 5;
  const int lane = threadIdx.x & 31;
  const int m0   = blockIdx.x * 32;                 // batch rows (2 M-tiles)
  const int nh0  = blockIdx.y * 64 + wave * 16;     // wave's hidden cols

  // Stage this wave's Wh slab for k-chunk [k0, k0+KC): 4 gates x 16 rows x
  // 128 B; 8 lanes per row -> 16 async b128 instructions per chunk. Rows are
  // padded to 144 B so ds_load_b128 fragment reads are bank-conflict-free.
  auto stage = [&](int buf, int k0) {
    const int rq = lane >> 3;                       // row-in-quad 0..3
    const int cb = (lane & 7) * 16;                 // byte offset within row
#pragma unroll
    for (int g = 0; g < 4; ++g) {
#pragma unroll
      for (int r4 = 0; r4 < 16; r4 += 4) {
        const int r = r4 + rq;                      // row 0..15 in wave tile
        const unsigned long long gaddr =
            (unsigned long long)WhT +
            (((unsigned long long)(g * H + nh0 + r) * H + (unsigned)k0) << 1) + cb;
        const int lrow = g * 64 + wave * 16 + r;
        const unsigned ldsoff =
            (unsigned)(size_t)(const void*)smem +
            (unsigned)(((buf * BROWS + lrow) * RSB) * 2 + cb);
        asm volatile("global_load_async_to_lds_b128 %0, %1, off"
                     :: "v"(ldsoff), "v"(gaddr) : "memory");
      }
    }
  };

  // Read one gate's 16x32 B fragment out of LDS (ds_load_b128 x2 per lane).
  auto load_b_lds = [&](int buf, int g, int kk) -> v16bf {
    const unsigned short* p =
        smem + (size_t)((buf * BROWS + g * 64 + wave * 16 + (lane & 15)) * RSB) +
        (size_t)(kk + ((lane >> 4) << 3));
    u32x4 lo = *(const u32x4*)p;
    u32x4 hi = *(const u32x4*)(p + 16);
    return pack_frag(lo, hi);
  };

  v8f z = {0.f,0.f,0.f,0.f,0.f,0.f,0.f,0.f};
  v8f aI0 = z, aI1 = z, aF0 = z, aF1 = z, aG0 = z, aG1 = z, aO0 = z, aO1 = z;

  stage(0, 0);                                      // prologue: chunk 0 DMA
  // Register-pipelined A fragments (k-step 0 preloaded).
  v16bf a0 = load_tile_bf16(hb_in, m0,      H, 0);
  v16bf a1 = load_tile_bf16(hb_in, m0 + 16, H, 0);

  constexpr int NCH = H / KC;                       // 16 chunks
  for (int ch = 0; ch < NCH; ++ch) {
    const int k0  = ch * KC;
    const int buf = ch & 1;
    if (ch + 1 < NCH) {
      // ds reads of the buffer we are about to overwrite completed last
      // iteration; make it explicit, launch the next chunk's DMA, then wait
      // only for the OLD 16 async ops (the fresh 16 hide behind the WMMAs).
      asm volatile("s_wait_dscnt 0x0" ::: "memory");
      stage(buf ^ 1, k0 + KC);
      asm volatile("s_wait_asynccnt 0x10" ::: "memory");
    } else {
      asm volatile("s_wait_asynccnt 0x0" ::: "memory");
    }
#pragma unroll
    for (int kk = 0; kk < KC; kk += 32) {
      const int knext = k0 + kk + 32;
      v16bf a0n, a1n;
      if (knext < H) {                              // prefetch next A k-step
        a0n = load_tile_bf16(hb_in, m0,      H, knext);
        a1n = load_tile_bf16(hb_in, m0 + 16, H, knext);
      }
      v16bf bI = load_b_lds(buf, 0, kk);
      v16bf bF = load_b_lds(buf, 1, kk);
      v16bf bG = load_b_lds(buf, 2, kk);
      v16bf bO = load_b_lds(buf, 3, kk);
      aI0 = wmma_bf16(a0, bI, aI0);  aI1 = wmma_bf16(a1, bI, aI1);
      aF0 = wmma_bf16(a0, bF, aF0);  aF1 = wmma_bf16(a1, bF, aF1);
      aG0 = wmma_bf16(a0, bG, aG0);  aG1 = wmma_bf16(a1, bG, aG1);
      aO0 = wmma_bf16(a0, bO, aO0);  aO1 = wmma_bf16(a1, bO, aO1);
      if (knext < H) { a0 = a0n; a1 = a1n; }
    }
  }

  // ---- LSTM nonlinearity in-register on the WMMA C/D layout ---------------
  const int ncol = nh0 + (lane & 15);
  const int mb   = (lane >> 4) * 8;
  auto epi = [&](int mt, v8f aI, v8f aF, v8f aG, v8f aO) {
#pragma unroll
    for (int v = 0; v < 8; ++v) {
      const int b_ = m0 + mt * 16 + mb + v;
      const size_t pj = ((size_t)t * B + b_) * NG;
      const float xi = aI[v] + bf2f(xprojb[pj + 0 * H + ncol]);
      const float xf = aF[v] + bf2f(xprojb[pj + 1 * H + ncol]);
      const float xg = aG[v] + bf2f(xprojb[pj + 2 * H + ncol]);
      const float xo = aO[v] + bf2f(xprojb[pj + 3 * H + ncol]);
      const float i_t = sigmoid_f(xi);
      const float f_t = sigmoid_f(xf);
      const float g_t = tanh_f(xg);
      const float o_t = sigmoid_f(xo);
      const size_t ci = (size_t)b_ * H + ncol;
      const float cn = g_t * i_t + c[ci] * f_t;
      c[ci] = cn;
      hb_out[ci] = f2bf(tanh_f(cn) * o_t);
    }
  };
  epi(0, aI0, aF0, aG0, aO0);
  epi(1, aI1, aF1, aG1, aO1);
}

// ---------------------------------------------------------------------------
// Kernel 6: final projection h[512,1024] @ Wph -> out f32 [512,128],
// software-pipelined like the input GEMM.
// ---------------------------------------------------------------------------
__global__ void __launch_bounds__(32)
final_proj_kernel(const unsigned short* __restrict__ hb,
                  const unsigned short* __restrict__ WpT,
                  const float* __restrict__ bp, float* __restrict__ out) {
  const int m0 = blockIdx.x * 16;
  const int n0 = blockIdx.y * 16;
  v8f acc = {0.f,0.f,0.f,0.f,0.f,0.f,0.f,0.f};
  v16bf a = load_tile_bf16(hb,  m0, H, 0);
  v16bf b = load_tile_bf16(WpT, n0, H, 0);
  for (int k0 = 32; k0 < H; k0 += 32) {
    v16bf an = load_tile_bf16(hb,  m0, H, k0);
    v16bf bn = load_tile_bf16(WpT, n0, H, k0);
    acc = wmma_bf16(a, b, acc);
    a = an; b = bn;
  }
  acc = wmma_bf16(a, b, acc);
  const int lane = threadIdx.x & 31;
  const int ncol = n0 + (lane & 15);
  const int mb   = (lane >> 4) * 8;
#pragma unroll
  for (int v = 0; v < 8; ++v) {
    const int b_ = m0 + mb + v;
    out[(size_t)b_ * NC + ncol] = acc[v] + bp[ncol];
  }
}

// ---------------------------------------------------------------------------
extern "C" void kernel_launch(void* const* d_in, const int* in_sizes, int n_in,
                              void* d_out, int out_size, void* d_ws, size_t ws_size,
                              hipStream_t stream) {
  (void)in_sizes; (void)n_in; (void)out_size; (void)ws_size;
  // setup_inputs() order: x, Wgx, Wgh, bg, Wix, Wih, bi, Wfx, Wfh, bf,
  //                       Wox, Woh, bo, Wph, bp
  const float* x   = (const float*)d_in[0];
  const float* Wgx = (const float*)d_in[1];
  const float* Wgh = (const float*)d_in[2];
  const float* bg  = (const float*)d_in[3];
  const float* Wix = (const float*)d_in[4];
  const float* Wih = (const float*)d_in[5];
  const float* bi  = (const float*)d_in[6];
  const float* Wfx = (const float*)d_in[7];
  const float* Wfh = (const float*)d_in[8];
  const float* bfv = (const float*)d_in[9];
  const float* Wox = (const float*)d_in[10];
  const float* Woh = (const float*)d_in[11];
  const float* bo  = (const float*)d_in[12];
  const float* Wph = (const float*)d_in[13];
  const float* bp  = (const float*)d_in[14];

  // Workspace carve-up (256B aligned).
  char* ws = (char*)d_ws;
  size_t off = 0;
  auto carve = [&](size_t bytes) {
    void* p = ws + off;
    off = (off + bytes + 255) & ~(size_t)255;
    return p;
  };
  unsigned short* WxT    = (unsigned short*)carve((size_t)NG * D * 2);     //   4 MB
  unsigned short* WhT    = (unsigned short*)carve((size_t)NG * H * 2);     //   8 MB
  unsigned short* WpT    = (unsigned short*)carve((size_t)NC * H * 2);     // 256 KB
  float*          bias   = (float*)carve((size_t)NG * 4);                  //  16 KB
  unsigned short* xb     = (unsigned short*)carve((size_t)BT * D * 2);     // 134 MB
  unsigned short* xprojb = (unsigned short*)carve((size_t)T * B * NG * 2); // 1.07 GB
  float*          cbuf   = (float*)carve((size_t)B * H * 4);               //   2 MB
  unsigned short* hbuf   = (unsigned short*)carve((size_t)2 * B * H * 2);  //   2 MB

  pack_weights_kernel<<<4096, 256, 0, stream>>>(
      Wix, Wfx, Wgx, Wox, Wih, Wfh, Wgh, Woh, bi, bfv, bg, bo, Wph,
      WxT, WhT, WpT, bias);
  pack_x_kernel<<<4096, 256, 0, stream>>>(x, xb, (long)BT * D / 4);
  init_state_kernel<<<1024, 256, 0, stream>>>(cbuf, hbuf, B * H);

  input_gemm_kernel<<<dim3(BT / 16, 16), 128, 0, stream>>>(xb, WxT, bias, xprojb);

  // Sequential recurrence: ping-pong h buffers (step t reads t&1, writes (t+1)&1).
  for (int t = 0; t < T; ++t) {
    unsigned short* h_in  = hbuf + (size_t)(t & 1) * B * H;
    unsigned short* h_out = hbuf + (size_t)((t + 1) & 1) * B * H;
    lstm_step_kernel<<<dim3(B / 32, H / 64), 128, STEP_LDS, stream>>>(
        WhT, xprojb, h_in, h_out, cbuf, t);
  }

  // T=256 is even -> final h lives in buffer 0.
  final_proj_kernel<<<dim3(B / 16, NC / 16), 32, 0, stream>>>(
      hbuf, WpT, bp, (float*)d_out);
}
